// STSGConv_31894427140604
// MI455X (gfx1250) — matrix operand order repaired and checked by reading, not compiled, not run on hardware
//
#include <hip/hip_runtime.h>

#define N_NODES 50000
#define N_EDGES 1600000
#define HID     128
#define OUTF    64

typedef __attribute__((ext_vector_type(16))) __bf16 v16bf;
typedef __attribute__((ext_vector_type(8)))  __bf16 v8bf;
typedef __attribute__((ext_vector_type(8)))  float  v8f;

static __device__ __forceinline__ __bf16 f2bf(float f) {
    union { float f; unsigned u; } c; c.f = f;
    unsigned r = c.u + 0x7FFFu + ((c.u >> 16) & 1u);   // round-to-nearest-even
    unsigned short h = (unsigned short)(r >> 16);
    return __builtin_bit_cast(__bf16, h);
}

// ---------------- graph-norm kernels ----------------

__global__ void k_zero(float* __restrict__ p, int n) {
    int t = blockIdx.x * blockDim.x + threadIdx.x;
    if (t < n) p[t] = 0.0f;
}

__global__ void k_degree(const int* __restrict__ col, float* __restrict__ deg, int e) {
    int t = blockIdx.x * blockDim.x + threadIdx.x;
    if (t < e) atomicAdd(&deg[col[t]], 1.0f);
}

__global__ void k_dinv(float* __restrict__ d, int n) {
    int t = blockIdx.x * blockDim.x + threadIdx.x;
    if (t < n) d[t] = rsqrtf(d[t] + 1.0f);
}

__global__ void k_enorm(const int* __restrict__ row, const int* __restrict__ col,
                        const float* __restrict__ dinv, float* __restrict__ enorm, int e) {
    int t = blockIdx.x * blockDim.x + threadIdx.x;
    if (t < e) enorm[t] = dinv[row[t]] * dinv[col[t]];
}

// ---------------- propagation: y = snorm*x + scatter_add(enorm*x[row] -> col) ----------------

__global__ void k_selfinit(const float* __restrict__ src, const float* __restrict__ dinv,
                           float* __restrict__ dst) {
    int t = blockIdx.x * blockDim.x + threadIdx.x;   // grid sized exactly N*HID
    float di = dinv[t >> 7];
    dst[t] = di * di * src[t];
}

__global__ void k_scatter(const float* __restrict__ src, const int* __restrict__ row,
                          const int* __restrict__ col, const float* __restrict__ enorm,
                          float* __restrict__ dst) {
    long long t = (long long)blockIdx.x * blockDim.x + threadIdx.x;  // grid sized exactly E*HID
    int e = (int)(t >> 7);
    int f = (int)(t & 127);
    float v = enorm[e] * src[(size_t)row[e] * HID + f];
    atomicAdd(&dst[(size_t)col[e] * HID + f], v);
}

// ---------------- f32 -> bf16 convert ----------------

__global__ void k_f2bf(const float* __restrict__ s, __bf16* __restrict__ d, int n) {
    int t = blockIdx.x * blockDim.x + threadIdx.x;
    if (t < n) d[t] = f2bf(s[t]);
}

// ---------------- WMMA GEMM: Y[N, nout] = X[N,128] @ W[nout,128]^T + bias (opt ReLU) ----------
// blockIdx.x = 16-row tile; wave id = 16-col tile. The block's 16x128 A tile is a contiguous
// 4 KB span of X -> cooperatively staged to LDS once, then each wave pulls WMMA fragments
// from LDS (ds path) instead of 8 redundant global re-reads.
// Epilogue writes fp32 (Yf) and/or bf16 (Ybf) so GEMM->GEMM chains skip fp32 round-trips.

__global__ void k_gemm_wmma(const __bf16* __restrict__ X, const __bf16* __restrict__ W,
                            const float* __restrict__ bias,
                            float* __restrict__ Yf, __bf16* __restrict__ Ybf,
                            int nout, int dorelu) {
    __shared__ __bf16 tileA[16 * HID];          // 4 KB
    const int K = HID;
    int tid  = threadIdx.x;
    int wave = tid >> 5;
    int lane = tid & 31;
    int half = lane >> 4;          // 0: lanes 0-15, 1: lanes 16-31
    int m    = lane & 15;

    // cooperative stage: 16 rows x 256 B = 4096 B contiguous in global
    {
        const uint4* src = (const uint4*)(X + (size_t)blockIdx.x * 16 * K);
        uint4*       dst = (uint4*)tileA;
        for (int i = tid; i < (16 * K) / 8; i += blockDim.x)   // 256 x 16B
            dst[i] = src[i];
    }
    __syncthreads();

    const __bf16* ap = tileA + m * K;                      // A: row M = lane%16 (from LDS)
    const __bf16* wp = W + (size_t)(wave * 16 + m) * K;    // B: col N = lane%16 = row of W

    v8f acc = {0.f, 0.f, 0.f, 0.f, 0.f, 0.f, 0.f, 0.f};
#pragma unroll
    for (int k0 = 0; k0 < K; k0 += 32) {
        // A 16x32 bf16 layout: lane holds 16 contiguous K values at half*16
        v8bf alo = *(const v8bf*)(ap + k0 + half * 16);
        v8bf ahi = *(const v8bf*)(ap + k0 + half * 16 + 8);
        v16bf a;
#pragma unroll
        for (int i = 0; i < 8; ++i) { a[i] = alo[i]; a[i + 8] = ahi[i]; }
        // B 32x16 bf16 layout: lane holds 16 contiguous K values at half*16
        v16bf b = *(const v16bf*)(wp + k0 + half * 16);
        acc = __builtin_amdgcn_wmma_f32_16x16x32_bf16(
            /*neg_a=*/false, a, /*neg_b=*/false, b,
            /*c_mod=*/(short)0, acc, /*reuse_a=*/false, /*reuse_b=*/false);
    }

    int ncol = wave * 16 + m;          // D: N = lane%16
    float bb = bias[ncol];
#pragma unroll
    for (int r = 0; r < 8; ++r) {      // D: M = r + 8*half
        float v = acc[r] + bb;
        if (dorelu) v = fmaxf(v, 0.0f);
        size_t idx = (size_t)(blockIdx.x * 16 + r + half * 8) * nout + ncol;
        if (Yf)  Yf[idx]  = v;
        if (Ybf) Ybf[idx] = f2bf(v);
    }
}

// ---------------- host-side orchestration ----------------

extern "C" void kernel_launch(void* const* d_in, const int* in_sizes, int n_in,
                              void* d_out, int out_size, void* d_ws, size_t ws_size,
                              hipStream_t stream) {
    (void)in_sizes; (void)n_in; (void)out_size; (void)ws_size;

    const float* x    = (const float*)d_in[0];
    const int*   ei   = (const int*)d_in[1];
    // d_in[2] = batch (unused: single graph, output is per-node)
    const float* W1   = (const float*)d_in[3];
    const float* b1   = (const float*)d_in[4];
    const float* W2   = (const float*)d_in[5];
    const float* b2   = (const float*)d_in[6];
    const float* Wqkv = (const float*)d_in[7];
    const float* bqkv = (const float*)d_in[8];
    const float* Wo   = (const float*)d_in[9];
    const float* bo   = (const float*)d_in[10];
    const float* Wl   = (const float*)d_in[11];
    const float* bl   = (const float*)d_in[12];
    float* out = (float*)d_out;

    // workspace carving (256B aligned blocks)
    char* wsp = (char*)d_ws;
    size_t off = 0;
    auto alloc = [&](size_t bytes) -> char* {
        off = (off + 255) & ~(size_t)255;
        char* p = wsp + off;
        off += bytes;
        return p;
    };
    float*  dinv   = (float*) alloc(sizeof(float) * N_NODES);
    float*  enorm  = (float*) alloc(sizeof(float) * N_EDGES);
    float*  bufA   = (float*) alloc(sizeof(float) * (size_t)N_NODES * HID);
    float*  bufB   = (float*) alloc(sizeof(float) * (size_t)N_NODES * HID);
    __bf16* bufBF  = (__bf16*)alloc(sizeof(__bf16) * (size_t)N_NODES * HID);
    __bf16* bufBF2 = (__bf16*)alloc(sizeof(__bf16) * (size_t)N_NODES * HID);
    __bf16* w1bf   = (__bf16*)alloc(sizeof(__bf16) * HID * HID);
    __bf16* w2bf   = (__bf16*)alloc(sizeof(__bf16) * HID * HID);
    __bf16* wvbf   = (__bf16*)alloc(sizeof(__bf16) * HID * HID);
    __bf16* wobf   = (__bf16*)alloc(sizeof(__bf16) * HID * HID);
    __bf16* wlbf   = (__bf16*)alloc(sizeof(__bf16) * OUTF * HID);

    const int* row = ei;
    const int* col = ei + N_EDGES;

    // GCN normalization
    k_zero  <<<(N_NODES + 255) / 256, 256, 0, stream>>>(dinv, N_NODES);
    k_degree<<<(N_EDGES + 255) / 256, 256, 0, stream>>>(col, dinv, N_EDGES);
    k_dinv  <<<(N_NODES + 255) / 256, 256, 0, stream>>>(dinv, N_NODES);
    k_enorm <<<(N_EDGES + 255) / 256, 256, 0, stream>>>(row, col, dinv, enorm, N_EDGES);

    // weight conversion (MHA with seq_len=1 collapses to the V projection: Wqkv rows [256,384))
    k_f2bf<<<(HID * HID + 255) / 256, 256, 0, stream>>>(W1, w1bf, HID * HID);
    k_f2bf<<<(HID * HID + 255) / 256, 256, 0, stream>>>(W2, w2bf, HID * HID);
    k_f2bf<<<(HID * HID + 255) / 256, 256, 0, stream>>>(Wqkv + 2 * HID * HID, wvbf, HID * HID);
    k_f2bf<<<(HID * HID + 255) / 256, 256, 0, stream>>>(Wo, wobf, HID * HID);
    k_f2bf<<<(OUTF * HID + 255) / 256, 256, 0, stream>>>(Wl, wlbf, OUTF * HID);

    const int nodeBlocks = (N_NODES * HID) / 256;   // 25000
    const int edgeBlocks = (N_EDGES * HID) / 256;   // 800000
    const int gemmGrid   = N_NODES / 16;            // 3125
    auto prop = [&](const float* s, float* d) {
        k_selfinit<<<nodeBlocks, 256, 0, stream>>>(s, dinv, d);
        k_scatter <<<edgeBlocks, 256, 0, stream>>>(s, row, col, enorm, d);
    };

    // layer 1: K=3 propagation, then relu(X @ W1^T + b1) -> fp32 (feeds propagation)
    prop(x, bufA); prop(bufA, bufB); prop(bufB, bufA);
    k_f2bf<<<nodeBlocks, 256, 0, stream>>>(bufA, bufBF, N_NODES * HID);
    k_gemm_wmma<<<gemmGrid, 256, 0, stream>>>(bufBF, w1bf, b1, bufB, (__bf16*)nullptr, HID, 1);

    // layer 2: propagation, then relu GEMM -> bf16 directly (feeds next GEMM)
    prop(bufB, bufA); prop(bufA, bufB); prop(bufB, bufA);
    k_f2bf<<<nodeBlocks, 256, 0, stream>>>(bufA, bufBF, N_NODES * HID);
    k_gemm_wmma<<<gemmGrid, 256, 0, stream>>>(bufBF, w2bf, b2, (float*)nullptr, bufBF2, HID, 1);

    // MHA (softmax over seq_len=1 is identity): v = h @ Wv^T + bv; h = v @ Wo^T + bo
    k_gemm_wmma<<<gemmGrid, 256, 0, stream>>>(bufBF2, wvbf, bqkv + 2 * HID, (float*)nullptr, bufBF, HID, 0);
    k_gemm_wmma<<<gemmGrid, 256, 0, stream>>>(bufBF, wobf, bo, (float*)nullptr, bufBF2, HID, 0);

    // final linear to [N, 64] -> d_out (4 column tiles -> 4 waves/block)
    k_gemm_wmma<<<gemmGrid, 128, 0, stream>>>(bufBF2, wlbf, bl, out, (__bf16*)nullptr, OUTF, 0);
}
